// FullGN_63694365000381
// MI455X (gfx1250) — matrix operand order repaired and checked by reading, not compiled, not run on hardware
//
#include <hip/hip_runtime.h>

#define DEV __device__ __forceinline__

typedef __attribute__((ext_vector_type(16))) __bf16 v16bf;
typedef __attribute__((ext_vector_type(2)))  __bf16 v2bf;
typedef __attribute__((ext_vector_type(8)))  float  v8f;
typedef __attribute__((ext_vector_type(4)))  float  v4f;
typedef __attribute__((ext_vector_type(4)))  unsigned u4;

constexpr int D  = 128;
constexpr int NN = 40000;
constexpr int EE = 640000;
constexpr int GG = 64;

constexpr int LROW = 68;          // LDS row stride in dwords (136 bf16): 16B aligned,
                                  // start-of-row bank = 4*r mod 64 -> conflict-free
constexpr int LTILE = 16 * LROW;  // dwords per staged 16x128 bf16 tile

union V16 { u4 q[2]; v16bf v; };

// ---- hardware bf16 pack (v_cvt_pk_bf16_f32-class codegen) ----
DEV unsigned pack2bf(float x, float y) {
  v2bf p;
  p[0] = (__bf16)x;
  p[1] = (__bf16)y;
  return __builtin_bit_cast(unsigned, p);
}

// ---- stage one 16x128 fp32 tile into LDS as bf16 (all 256 threads) ----
template <bool NT>
DEV void stage_tile(const float* __restrict__ X, int m0, unsigned* lds) {
  const int row = threadIdx.x >> 4;         // 0..15
  const int c   = (threadIdx.x & 15) << 3;  // 0,8,...,120
  const float* src = X + (size_t)(m0 + row) * D + c;
  v4f x0, x1;
  if (NT) {
    x0 = __builtin_nontemporal_load((const v4f*)src);
    x1 = __builtin_nontemporal_load((const v4f*)src + 1);
  } else {
    x0 = *(const v4f*)src;
    x1 = *((const v4f*)src + 1);
  }
  u4 p;
  p[0] = pack2bf(x0[0], x0[1]);
  p[1] = pack2bf(x0[2], x0[3]);
  p[2] = pack2bf(x1[0], x1[1]);
  p[3] = pack2bf(x1[2], x1[3]);
  *(u4*)(lds + row * LROW + (c >> 1)) = p;
}

// ---- one 16x16 tile of  acc += A(LDS bf16 16x128) @ W[:, n0:n0+16] ----
// A fragment (16x32): lane r=lane%16 -> M; h=lane/16; vgpr v -> K =
//   kb*32 + (v/4)*16 + h*8 + (v%4)*2  => dwords kb*16 + h*4 + {0..3} and +8
// B fragment (32x16): lane r -> N; vgpr v -> K = kb*32 + (v/4)*16 + h*8 + (v%4)*2 (same as A)
//   => dwords row(n)*64 + kb*16 + h*4 + {0..3} and +8
// C/D: vgpr v -> M = h*8 + v, lane r -> N
DEV v8f tile_gemm(const unsigned* __restrict__ lA,
                  const unsigned short* __restrict__ WT, int n0, v8f acc) {
  const int lane = threadIdx.x & 31;
  const int r = lane & 15, h = lane >> 4;
  const unsigned* arow = lA + r * LROW + h * 4;
  const unsigned* wrow = (const unsigned*)(WT + (size_t)(n0 + r) * D) + h * 4;
#pragma unroll
  for (int kb = 0; kb < 4; ++kb) {
    V16 A, B;
    A.q[0] = *(const u4*)(arow + kb * 16);
    A.q[1] = *(const u4*)(arow + kb * 16 + 8);
    B.q[0] = *(const u4*)(wrow + kb * 16);
    B.q[1] = *(const u4*)(wrow + kb * 16 + 8);
    acc = __builtin_amdgcn_wmma_f32_16x16x32_bf16(
        /*neg_a=*/false, A.v, /*neg_b=*/false, B.v,
        /*c_mod=*/(short)0, acc, /*reuse_a=*/false, /*reuse_b=*/false);
  }
  return acc;
}

// ---- convert one DxD fp32 weight to bf16 transposed ----
__global__ void __launch_bounds__(256) k_wtrans(const float* __restrict__ W,
                                                unsigned short* __restrict__ WT) {
  int idx = blockIdx.x * 256 + threadIdx.x;  // 0..16383
  int n = idx >> 7, k = idx & 127;
  WT[idx] = __builtin_bit_cast(unsigned short, (__bf16)W[k * D + n]);
}

// ---- Y = X @ W + bias  (M % 16 == 0) ----
__global__ void __launch_bounds__(256) k_gemm_bias(const float* __restrict__ X,
                                                   const unsigned short* __restrict__ WT,
                                                   const float* __restrict__ bias,
                                                   float* __restrict__ Y) {
  __shared__ unsigned lA[LTILE];
  const int m0 = blockIdx.x * 16;
  stage_tile<false>(X, m0, lA);
  __syncthreads();
  const int n0 = (threadIdx.x >> 5) * 16;
  v8f acc = {};
  acc = tile_gemm(lA, WT, n0, acc);
  const int lane = threadIdx.x & 31;
  const int r = lane & 15, h = lane >> 4;
  const int n = n0 + r;
  const float bv = bias[n];
#pragma unroll
  for (int v = 0; v < 8; ++v) {
    int m = m0 + h * 8 + v;
    Y[(size_t)m * D + n] = acc[v] + bv;
  }
}

// ---- edge update: GEMM + gathers + ReLU + fused segment-sum scatters ----
__global__ void __launch_bounds__(256) k_edge(
    const float* __restrict__ EF,
    const int* __restrict__ snd, const int* __restrict__ rcv,
    const int* __restrict__ egid,
    const unsigned short* __restrict__ WTe, const float* __restrict__ be,
    const float* __restrict__ Ps, const float* __restrict__ Pr,
    const float* __restrict__ Pue,
    float* __restrict__ out_edges, float* __restrict__ Ain,
    float* __restrict__ Aout, float* __restrict__ Gedge) {
  __shared__ unsigned lA[LTILE];
  const int m0 = blockIdx.x * 16;
  stage_tile<true>(EF, m0, lA);  // streamed once: nontemporal, keep L2 for gathers
  __syncthreads();
  const int n0 = (threadIdx.x >> 5) * 16;
  v8f acc = {};
  acc = tile_gemm(lA, WTe, n0, acc);
  const int lane = threadIdx.x & 31;
  const int r = lane & 15, h = lane >> 4;
  const int n = n0 + r;
  const float bv = be[n];
  float gsum = 0.f;
  int gprev = egid[m0 + h * 8];
#pragma unroll
  for (int v = 0; v < 8; ++v) {
    const int m = m0 + h * 8 + v;
    const int s = snd[m], rc = rcv[m], g = egid[m];
    float val = acc[v] + bv + Ps[(size_t)s * D + n] + Pr[(size_t)rc * D + n] +
                Pue[g * D + n];
    val = fmaxf(val, 0.f);
    __builtin_nontemporal_store(val, &out_edges[(size_t)m * D + n]);
    atomicAdd(&Ain[(size_t)rc * D + n], val);
    atomicAdd(&Aout[(size_t)s * D + n], val);
    if (g == gprev) {
      gsum += val;
    } else {
      atomicAdd(&Gedge[gprev * D + n], gsum);
      gprev = g;
      gsum = val;
    }
  }
  atomicAdd(&Gedge[gprev * D + n], gsum);
}

// ---- node update: three chained WMMA GEMMs + ReLU + graph node sums ----
__global__ void __launch_bounds__(256) k_node(
    const float* __restrict__ NF, const float* __restrict__ Ain,
    const float* __restrict__ Aout, const int* __restrict__ ngid,
    const unsigned short* __restrict__ WTn, const unsigned short* __restrict__ WTi,
    const unsigned short* __restrict__ WTo,
    const float* __restrict__ bn, const float* __restrict__ bi,
    const float* __restrict__ bo, const float* __restrict__ Pun,
    float* __restrict__ out_nodes, float* __restrict__ Gnode) {
  __shared__ unsigned lN[LTILE], lI[LTILE], lO[LTILE];
  const int m0 = blockIdx.x * 16;
  stage_tile<false>(NF, m0, lN);
  stage_tile<false>(Ain, m0, lI);
  stage_tile<false>(Aout, m0, lO);
  __syncthreads();
  const int n0 = (threadIdx.x >> 5) * 16;
  v8f acc = {};
  acc = tile_gemm(lN, WTn, n0, acc);
  acc = tile_gemm(lI, WTi, n0, acc);
  acc = tile_gemm(lO, WTo, n0, acc);
  const int lane = threadIdx.x & 31;
  const int r = lane & 15, h = lane >> 4;
  const int n = n0 + r;
  const float bv = bn[n] + bi[n] + bo[n];
  float gsum = 0.f;
  int gprev = ngid[m0 + h * 8];
#pragma unroll
  for (int v = 0; v < 8; ++v) {
    const int m = m0 + h * 8 + v;
    const int g = ngid[m];
    float val = acc[v] + bv + Pun[g * D + n];
    val = fmaxf(val, 0.f);
    out_nodes[(size_t)m * D + n] = val;
    if (g == gprev) {
      gsum += val;
    } else {
      atomicAdd(&Gnode[gprev * D + n], gsum);
      gprev = g;
      gsum = val;
    }
  }
  atomicAdd(&Gnode[gprev * D + n], gsum);
}

// ---- global update ----
__global__ void __launch_bounds__(256) k_global(
    const float* __restrict__ GF, const float* __restrict__ Gedge,
    const float* __restrict__ Gnode,
    const unsigned short* __restrict__ WThe, const unsigned short* __restrict__ WThn,
    const unsigned short* __restrict__ WThu,
    const float* __restrict__ bhe, const float* __restrict__ bhn,
    const float* __restrict__ bhu, float* __restrict__ out_g) {
  __shared__ unsigned lE[LTILE], lN[LTILE], lU[LTILE];
  const int m0 = blockIdx.x * 16;
  stage_tile<false>(Gedge, m0, lE);
  stage_tile<false>(Gnode, m0, lN);
  stage_tile<false>(GF, m0, lU);
  __syncthreads();
  const int n0 = (threadIdx.x >> 5) * 16;
  v8f acc = {};
  acc = tile_gemm(lE, WThe, n0, acc);
  acc = tile_gemm(lN, WThn, n0, acc);
  acc = tile_gemm(lU, WThu, n0, acc);
  const int lane = threadIdx.x & 31;
  const int r = lane & 15, h = lane >> 4;
  const int n = n0 + r;
  const float bv = bhe[n] + bhn[n] + bhu[n];
#pragma unroll
  for (int v = 0; v < 8; ++v) {
    int m = m0 + h * 8 + v;
    out_g[(size_t)m * D + n] = acc[v] + bv;
  }
}

extern "C" void kernel_launch(void* const* d_in, const int* in_sizes, int n_in,
                              void* d_out, int out_size, void* d_ws, size_t ws_size,
                              hipStream_t stream) {
  (void)in_sizes; (void)n_in; (void)out_size; (void)ws_size;
  const float* EF  = (const float*)d_in[0];
  const float* NF  = (const float*)d_in[1];
  const float* GF  = (const float*)d_in[2];
  const int* snd   = (const int*)d_in[3];
  const int* rcv   = (const int*)d_in[4];
  const int* egid  = (const int*)d_in[5];
  const int* ngid  = (const int*)d_in[6];
  const float* W[11];
  const float* B[11];
  // slot order: 0 We, 1 Ws, 2 Wr, 3 Wue, 4 Wn, 5 Wi, 6 Wo, 7 Wun, 8 Whn, 9 Whe, 10 Whu
  for (int i = 0; i < 11; ++i) {
    W[i] = (const float*)d_in[7 + 2 * i];
    B[i] = (const float*)d_in[8 + 2 * i];
  }

  char* ws = (char*)d_ws;
  unsigned short* WT = (unsigned short*)ws;          // 11 * 16384 bf16 = 360448 B
  float* Psend = (float*)(ws + 360448);              // 20,480,000 B
  float* Precv = (float*)(ws + 20840448);            // 20,480,000 B
  float* Pue   = (float*)(ws + 41320448);            // 32,768 B
  float* Pun   = (float*)(ws + 41353216);            // 32,768 B
  float* Ain   = (float*)(ws + 41385984);            // 20,480,000 B
  float* Aout  = (float*)(ws + 61865984);            // 20,480,000 B
  float* Gedge = (float*)(ws + 82345984);            // 32,768 B
  float* Gnode = (float*)(ws + 82378752);            // 32,768 B  (end 82,411,520)

  // zero the accumulators (Ain..Gnode are contiguous)
  (void)hipMemsetAsync(ws + 41385984, 0, 41025536, stream);

  // bf16-transpose all 11 weights (360448 B bf16 total, trivial)
  for (int i = 0; i < 11; ++i)
    k_wtrans<<<64, 256, 0, stream>>>(W[i], WT + (size_t)i * D * D);

  // node / global projections (L2-resident)
  k_gemm_bias<<<NN / 16, 256, 0, stream>>>(NF, WT + 1 * D * D, B[1], Psend);
  k_gemm_bias<<<NN / 16, 256, 0, stream>>>(NF, WT + 2 * D * D, B[2], Precv);
  k_gemm_bias<<<GG / 16, 256, 0, stream>>>(GF, WT + 3 * D * D, B[3], Pue);
  k_gemm_bias<<<GG / 16, 256, 0, stream>>>(GF, WT + 7 * D * D, B[7], Pun);

  float* out       = (float*)d_out;
  float* out_edges = out;
  float* out_nodes = out + (size_t)EE * D;
  float* out_glob  = out + (size_t)EE * D + (size_t)NN * D;

  k_edge<<<EE / 16, 256, 0, stream>>>(EF, snd, rcv, egid, WT /*We*/, B[0],
                                      Psend, Precv, Pue, out_edges, Ain, Aout,
                                      Gedge);
  k_node<<<NN / 16, 256, 0, stream>>>(NF, Ain, Aout, ngid, WT + 4 * D * D,
                                      WT + 5 * D * D, WT + 6 * D * D, B[4],
                                      B[5], B[6], Pun, out_nodes, Gnode);
  k_global<<<GG / 16, 256, 0, stream>>>(GF, Gedge, Gnode, WT + 9 * D * D,
                                        WT + 8 * D * D, WT + 10 * D * D, B[9],
                                        B[8], B[10], out_glob);
}
